// MSDeformAttnPixelDecoder_89962384982822
// MI455X (gfx1250) — compile-verified, matrix operand order checked by reading, not run on hardware
//
#include <hip/hip_runtime.h>
#include <hip/hip_bf16.h>

// ---------------- static problem dims ----------------
#define BATCH  4
#define CDIM   256
#define NHEAD  8
#define NLVL   3
#define NPT    4
#define NLAY   6
#define DFFN   1024
#define HDIM   32
#define SSZ    21504                 // 32*32 + 64*64 + 128*128
#define MROWS  (BATCH * SSZ)         // 86016 token rows (mult of 256)
#define AWPAD  128                   // Wa N=96 padded to 128 (16-aligned tiles)
#define OFFC   (NHEAD * NLVL * NPT * 2)   // 192
#define AWC    (NHEAD * NLVL * NPT)       // 96
#define PANEL_KT 8                   // K-tiles staged per panel (32KB LDS)

typedef __attribute__((ext_vector_type(16))) _Float16 v16h;
typedef __attribute__((ext_vector_type(8)))  _Float16 v8h;
typedef __attribute__((ext_vector_type(8)))  float    v8f;
typedef unsigned int u32x4 __attribute__((ext_vector_type(4)));
typedef int          i32x8 __attribute__((ext_vector_type(8)));
typedef int          i32x4 __attribute__((ext_vector_type(4)));

#if defined(__has_builtin)
#if __has_builtin(__builtin_amdgcn_tensor_load_to_lds) && \
    __has_builtin(__builtin_amdgcn_s_wait_tensorcnt)
#define MSDA_USE_TDM 1
#endif
#endif

// WMMA 16-bit fragment swizzle (ISA 7.12.2 layout):
//   element (m|n, k) -> tile (mt|nt = mn>>4, kt = k>>5),
//   lane = (mn&15) | ((k>>3)&1)<<4,  idx = (k&7) | ((k>>4)&1)<<3
// stored as [tile][lane][16 halves] => each lane's fragment is 32 contiguous B.
__device__ __forceinline__ long long swz_idx(int mn, int k, int Kdim) {
    const int lane = (mn & 15) | (((k >> 3) & 1) << 4);
    const int idx  = (k & 7) | (((k >> 4) & 1) << 3);
    const long long tile = (long long)(mn >> 4) * (Kdim >> 5) + (k >> 5);
    return ((tile << 5) + lane) * 16 + idx;
}

// ---------------- weight convert f32 -> f16 + B-fragment swizzle ----------
__global__ __launch_bounds__(256) void k_convert_swz(
    const float* __restrict__ src, _Float16* __restrict__ dst,
    int L, int K, int N, int Npad)
{
    long long i = (long long)blockIdx.x * 256 + threadIdx.x;
    const long long per = (long long)K * Npad;
    if (i >= (long long)L * per) return;
    const int  l    = (int)(i / per);
    const long long r = i % per;
    const int  idx  = (int)(r & 15);
    const int  lane = (int)((r >> 4) & 31);
    const long long tile = r >> 9;
    const int  NT   = Npad >> 4;
    const int  nt   = (int)(tile % NT);
    const int  kt   = (int)(tile / NT);
    const int  k    = (kt << 5) | (idx & 7) | ((lane >> 4) << 3) | ((idx >> 3) << 4);
    const int  n    = (nt << 4) | (lane & 15);
    const float v = (n < N) ? src[((long long)l * K + k) * N + n] : 0.0f;
    dst[i] = (_Float16)v;
}

// ---------------- flatten [B,C,H,W] -> [B,S,C]; pos += level_embed -------
__global__ __launch_bounds__(256) void k_flatten(
    const float* __restrict__ s0, const float* __restrict__ p0,
    const float* __restrict__ s1, const float* __restrict__ p1,
    const float* __restrict__ s2, const float* __restrict__ p2,
    const float* __restrict__ lemb,
    float* __restrict__ out_f, _Float16* __restrict__ out_h,
    float* __restrict__ pos_f, _Float16* __restrict__ q_h)
{
    long long i = (long long)blockIdx.x * 256 + threadIdx.x;
    if (i >= (long long)MROWS * CDIM) return;
    int c = (int)(i % CDIM);
    long long t = i / CDIM;
    int s = (int)(t % SSZ);
    int b = (int)(t / SSZ);
    int lvl, hw, HW;
    const float *sp, *pp;
    if (s < 1024)      { lvl = 0; hw = s;        HW = 1024;  sp = s0; pp = p0; }
    else if (s < 5120) { lvl = 1; hw = s - 1024; HW = 4096;  sp = s1; pp = p1; }
    else               { lvl = 2; hw = s - 5120; HW = 16384; sp = s2; pp = p2; }
    long long si = ((long long)b * CDIM + c) * HW + hw;
    float sv = sp[si];
    float pv = pp[si] + lemb[lvl * CDIM + c];
    const long long sw = swz_idx((int)t, c, CDIM);
    out_f[i] = sv;
    out_h[sw] = (_Float16)sv;
    pos_f[i] = pv;
    q_h[sw] = (_Float16)(sv + pv);     // layer-0 q fused here
}

// ---------------- WMMA GEMM on pre-swizzled operands ----------------------
// Asw: [M/16][K/32][32][16] f16 ; Bsw: [K/32][N/16][32][16] f16.
// block = 256 thr = 8 waves; block tile 256x64; wave tile 32x64.
// B staged in 8-K-tile (32KB) panels by the Tensor Data Mover (wave 0 issues
// tensor_load_to_lds, waits TENSORcnt, block barriers once per panel).
// EPI: 0 -> C f32 row-major ; 1 -> f16 ReLU, swizzled for next GEMM (Kout).
template<int EPI>
__global__ __launch_bounds__(256) void k_gemm_wmma(
    const _Float16* __restrict__ Asw, const _Float16* __restrict__ Bsw,
    const float* __restrict__ bias, int Nb,
    float* __restrict__ Cf, _Float16* __restrict__ Ch,
    int M, int N, int K, int Kout)
{
    __shared__ __align__(32) _Float16 lB[PANEL_KT * 2048];   // 8 K-tiles * 4KB

    const int tid  = threadIdx.x;
    const int wave = tid >> 5;
    const int lane = tid & 31;
    const int m0   = blockIdx.y * 256 + wave * 32;
    const int nt0  = blockIdx.x * 4;
    const int KT   = K >> 5;
    const int NT   = N >> 4;

    v8f acc[8];
    #pragma unroll
    for (int j = 0; j < 8; j++)
        #pragma unroll
        for (int e = 0; e < 8; e++) acc[j][e] = 0.0f;

    const _Float16* pa0 = Asw + ((((long long)(m0 >> 4))     * KT) << 9) + lane * 16;
    const _Float16* pa1 = Asw + ((((long long)(m0 >> 4) + 1) * KT) << 9) + lane * 16;
    const char* gpanel0 = (const char*)Bsw + ((long long)nt0 << 10);

    for (int p = 0; p < KT; p += PANEL_KT) {
        __syncthreads();   // previous panel fully consumed before overwrite
        const char* gB = gpanel0 + (((long long)p * NT) << 10);
#ifdef MSDA_USE_TDM
        if (wave == 0) {
            // D# for a 2D tile: rows of 4096B (one kt group), row stride NT KB.
            const unsigned long long ga = (unsigned long long)(uintptr_t)gB;
            const unsigned int lds_addr = (unsigned int)(uintptr_t)(&lB[0]);
            u32x4 g0 = {0u, 0u, 0u, 0u};
            g0[0] = 1u;                                          // count=1
            g0[1] = lds_addr;                                    // lds_addr
            g0[2] = (unsigned int)(ga & 0xFFFFFFFFu);            // global_addr lo
            g0[3] = (unsigned int)((ga >> 32) & 0x01FFFFFFu)     // global_addr hi
                    | (2u << 30);                                // type = 2 (image)
            i32x8 g1 = {0, 0, 0, 0, 0, 0, 0, 0};
            g1[0] = (3 << 16);                // data_size = 8 bytes
            g1[1] = (int)(512u << 16);        // tensor_dim0 = 512 (8B units)
            g1[2] = (int)((unsigned)PANEL_KT << 16);   // tensor_dim1 = 8
            g1[3] = (int)(512u << 16);        // tile_dim0 = 512
            g1[4] = PANEL_KT;                 // tile_dim1 = 8
            g1[5] = NT * 128;                 // tensor_dim0_stride (8B units)
            i32x4 gz4 = {0, 0, 0, 0};
            i32x8 gz8 = {0, 0, 0, 0, 0, 0, 0, 0};
            __builtin_amdgcn_tensor_load_to_lds(g0, g1, gz4, gz4, gz8, 0);
            __builtin_amdgcn_s_wait_tensorcnt(0);
        }
#else
        // fallback: cooperative coalesced copy, 16B per thread per step
        #pragma unroll
        for (int c = 0; c < 8; c++) {
            const int off = (c * 256 + tid) * 16;
            const int row = off >> 12, within = off & 4095;
            *(v8h*)((char*)lB + off) =
                *(const v8h*)(gB + (((long long)row * NT) << 10) + within);
        }
#endif
        __syncthreads();

        // 8 K-steps from LDS: no barriers, no staging in this loop
        #pragma unroll
        for (int j = 0; j < PANEL_KT; j++) {
            const int kt = p + j;
            const v16h a0 = *(const v16h*)(pa0 + ((long long)kt << 9));
            const v16h a1 = *(const v16h*)(pa1 + ((long long)kt << 9));
            #pragma unroll
            for (int t = 0; t < 4; t++) {
                const v16h b = *(const v16h*)(&lB[j * 2048 + t * 512 + lane * 16]);
                acc[t]     = __builtin_amdgcn_wmma_f32_16x16x32_f16(
                                 false, a0, false, b, (short)0, acc[t],     false, false);
                acc[4 + t] = __builtin_amdgcn_wmma_f32_16x16x32_f16(
                                 false, a1, false, b, (short)0, acc[4 + t], false, false);
            }
        }
    }

    // epilogue; C layout: VGPR r -> M = m0 + 16*half + (lane>=16 ? 8 : 0) + r
    const int rbase = (lane >> 4) * 8;
    #pragma unroll
    for (int half = 0; half < 2; half++) {
        #pragma unroll
        for (int j = 0; j < 4; j++) {
            const int col = (nt0 + j) * 16 + (lane & 15);
            const float bsv = (col < Nb) ? bias[col] : 0.0f;
            #pragma unroll
            for (int r = 0; r < 8; r++) {
                const int row = m0 + half * 16 + rbase + r;
                const float v = acc[half * 4 + j][r] + bsv;
                if (EPI == 0) {
                    Cf[(long long)row * N + col] = v;
                } else {
                    Ch[swz_idx(row, col, Kout)] = (_Float16)fmaxf(v, 0.0f);
                }
            }
        }
    }
}

// ---------------- fused softmax + bilinear deformable sampling ------------
__global__ __launch_bounds__(256) void k_sample(
    const float* __restrict__ value,   // [B*S, 256] row-major
    const float* __restrict__ offs,    // [B*S, 192] row-major
    const float* __restrict__ awl,     // [B*S, AWPAD] logits row-major
    _Float16* __restrict__ attn_h)     // swizzled [B*S, 256]
{
    const int wid  = blockIdx.x * 8 + (threadIdx.x >> 5);
    const int lane = threadIdx.x & 31;
    if (wid >= MROWS * NHEAD) return;
    const int h  = wid % NHEAD;
    const int bs = wid / NHEAD;
    const int s  = bs % SSZ;

    const float* lrow = awl + (long long)bs * AWPAD + h * (NLVL * NPT);
    float lg[12], mx = -1e30f;
    #pragma unroll
    for (int i = 0; i < 12; i++) { lg[i] = lrow[i]; mx = fmaxf(mx, lg[i]); }
    float den = 0.0f;
    #pragma unroll
    for (int i = 0; i < 12; i++) { lg[i] = __expf(lg[i] - mx); den += lg[i]; }
    const float rden = 1.0f / den;

    int WL0, li;
    if (s < 1024)      { WL0 = 32;  li = s; }
    else if (s < 5120) { WL0 = 64;  li = s - 1024; }
    else               { WL0 = 128; li = s - 5120; }
    const float rx = ((li % WL0) + 0.5f) / WL0;
    const float ry = ((li / WL0) + 0.5f) / WL0;   // square levels: H == W

    const float* orow = offs + (long long)bs * OFFC + h * (NLVL * NPT * 2);
    const int d = h * HDIM + lane;
    const long long vbatch = (long long)(bs / SSZ) * SSZ;

    const int LWH[3] = {32, 64, 128};
    const int LS[3]  = {0, 1024, 5120};

    float acc = 0.0f;
    #pragma unroll
    for (int lvl = 0; lvl < NLVL; lvl++) {
        const int WL = LWH[lvl], st = LS[lvl];
        #pragma unroll
        for (int p = 0; p < NPT; p++) {
            const float ox = orow[lvl * (NPT * 2) + p * 2 + 0];
            const float oy = orow[lvl * (NPT * 2) + p * 2 + 1];
            const float x = (rx + ox / WL) * WL - 0.5f;
            const float y = (ry + oy / WL) * WL - 0.5f;
            const float x0f = floorf(x), y0f = floorf(y);
            const int x0 = (int)x0f, y0 = (int)y0f;
            const float fx = x - x0f, fy = y - y0f;
            const float wgt = lg[lvl * NPT + p] * rden;

            float v00 = 0.f, v01 = 0.f, v10 = 0.f, v11 = 0.f;
            const bool xi0 = (x0 >= 0) & (x0 < WL);
            const bool xi1 = (x0 + 1 >= 0) & (x0 + 1 < WL);
            const bool yi0 = (y0 >= 0) & (y0 < WL);
            const bool yi1 = (y0 + 1 >= 0) & (y0 + 1 < WL);
            if (yi0 & xi0) v00 = value[(vbatch + st + (long long)y0 * WL + x0) * CDIM + d];
            if (yi0 & xi1) v01 = value[(vbatch + st + (long long)y0 * WL + x0 + 1) * CDIM + d];
            if (yi1 & xi0) v10 = value[(vbatch + st + (long long)(y0 + 1) * WL + x0) * CDIM + d];
            if (yi1 & xi1) v11 = value[(vbatch + st + (long long)(y0 + 1) * WL + x0 + 1) * CDIM + d];

            acc += wgt * (v00 * (1.f - fx) * (1.f - fy) + v01 * fx * (1.f - fy)
                        + v10 * (1.f - fx) * fy        + v11 * fx * fy);
        }
    }
    attn_h[swz_idx(bs, d, CDIM)] = (_Float16)acc;
}

// ---------------- residual add + LayerNorm (+ optional fused next-q) ------
__global__ __launch_bounds__(256) void k_resid_ln(
    float* __restrict__ out_f, const float* __restrict__ delta,
    const float* __restrict__ g, const float* __restrict__ bta,
    _Float16* __restrict__ out_h,
    const float* __restrict__ pos_f, _Float16* __restrict__ q_h)
{
    const int row  = blockIdx.x * 8 + (threadIdx.x >> 5);
    const int lane = threadIdx.x & 31;
    if (row >= MROWS) return;
    const long long base = (long long)row * CDIM;

    float v[8], sum = 0.0f;
    #pragma unroll
    for (int i = 0; i < 8; i++) {
        v[i] = out_f[base + lane + i * 32] + delta[base + lane + i * 32];
        sum += v[i];
    }
    #pragma unroll
    for (int off = 16; off > 0; off >>= 1) sum += __shfl_xor(sum, off, 32);
    const float mean = sum * (1.0f / CDIM);

    float vs = 0.0f;
    #pragma unroll
    for (int i = 0; i < 8; i++) { float c = v[i] - mean; vs += c * c; }
    #pragma unroll
    for (int off = 16; off > 0; off >>= 1) vs += __shfl_xor(vs, off, 32);
    const float inv = rsqrtf(vs * (1.0f / CDIM) + 1e-5f);

    #pragma unroll
    for (int i = 0; i < 8; i++) {
        const int c = lane + i * 32;
        const float y = (v[i] - mean) * inv * g[c] + bta[c];
        const long long sw = swz_idx(row, c, CDIM);
        out_f[base + c] = y;
        out_h[sw] = (_Float16)y;
        if (q_h) q_h[sw] = (_Float16)(y + pos_f[base + c]);  // next layer's q
    }
}

// ---------------------------------------------------------------------------
extern "C" void kernel_launch(void* const* d_in, const int* in_sizes, int n_in,
                              void* d_out, int out_size, void* d_ws, size_t ws_size,
                              hipStream_t stream) {
    (void)in_sizes; (void)n_in; (void)out_size; (void)ws_size;
    const float* src0 = (const float*)d_in[0];
    const float* pos0 = (const float*)d_in[1];
    const float* src1 = (const float*)d_in[2];
    const float* pos1 = (const float*)d_in[3];
    const float* src2 = (const float*)d_in[4];
    const float* pos2 = (const float*)d_in[5];
    const float* lemb = (const float*)d_in[6];
    const float* Wv   = (const float*)d_in[7];  const float* bv   = (const float*)d_in[8];
    const float* Wo   = (const float*)d_in[9];  const float* bo   = (const float*)d_in[10];
    const float* Wa   = (const float*)d_in[11]; const float* ba   = (const float*)d_in[12];
    const float* Wout = (const float*)d_in[13]; const float* bout = (const float*)d_in[14];
    const float* ln1g = (const float*)d_in[15]; const float* ln1b = (const float*)d_in[16];
    const float* W1   = (const float*)d_in[17]; const float* b1   = (const float*)d_in[18];
    const float* W2   = (const float*)d_in[19]; const float* b2   = (const float*)d_in[20];
    const float* ln2g = (const float*)d_in[21]; const float* ln2b = (const float*)d_in[22];

    char* wp = (char*)d_ws;
    auto carve = [&](size_t bytes) -> void* {
        void* p = (void*)wp; wp += (bytes + 255) & ~(size_t)255; return p;
    };
    const size_t MC = (size_t)MROWS * CDIM;
    float*     out_f   = (float*)    carve(MC * 4);
    float*     pos_f   = (float*)    carve(MC * 4);
    _Float16*  out_h   = (_Float16*) carve(MC * 2);
    _Float16*  q_h     = (_Float16*) carve(MC * 2);
    float*     val_f   = (float*)    carve(MC * 4);
    float*     offs_f  = (float*)    carve((size_t)MROWS * OFFC * 4);
    float*     awl_f   = (float*)    carve((size_t)MROWS * AWPAD * 4);
    _Float16*  attn_h  = (_Float16*) carve(MC * 2);
    float*     tmp_f   = (float*)    carve(MC * 4);
    _Float16*  hid_h   = (_Float16*) carve((size_t)MROWS * DFFN * 2);
    _Float16*  Wv16    = (_Float16*) carve((size_t)NLAY * CDIM * CDIM * 2);
    _Float16*  Wo16    = (_Float16*) carve((size_t)NLAY * CDIM * OFFC * 2);
    _Float16*  Wa16    = (_Float16*) carve((size_t)NLAY * CDIM * AWPAD * 2);
    _Float16*  Wout16  = (_Float16*) carve((size_t)NLAY * CDIM * CDIM * 2);
    _Float16*  W116    = (_Float16*) carve((size_t)NLAY * CDIM * DFFN * 2);
    _Float16*  W216    = (_Float16*) carve((size_t)NLAY * DFFN * CDIM * 2);

    auto cgrid = [](long long n) { return dim3((unsigned)((n + 255) / 256)); };

    k_convert_swz<<<cgrid((long long)NLAY * CDIM * CDIM),  256, 0, stream>>>(Wv,   Wv16,   NLAY, CDIM, CDIM, CDIM);
    k_convert_swz<<<cgrid((long long)NLAY * CDIM * OFFC),  256, 0, stream>>>(Wo,   Wo16,   NLAY, CDIM, OFFC, OFFC);
    k_convert_swz<<<cgrid((long long)NLAY * CDIM * AWPAD), 256, 0, stream>>>(Wa,   Wa16,   NLAY, CDIM, AWC,  AWPAD);
    k_convert_swz<<<cgrid((long long)NLAY * CDIM * CDIM),  256, 0, stream>>>(Wout, Wout16, NLAY, CDIM, CDIM, CDIM);
    k_convert_swz<<<cgrid((long long)NLAY * CDIM * DFFN),  256, 0, stream>>>(W1,   W116,   NLAY, CDIM, DFFN, DFFN);
    k_convert_swz<<<cgrid((long long)NLAY * DFFN * CDIM),  256, 0, stream>>>(W2,   W216,   NLAY, DFFN, CDIM, CDIM);

    k_flatten<<<cgrid((long long)MC), 256, 0, stream>>>(
        src0, pos0, src1, pos1, src2, pos2, lemb, out_f, out_h, pos_f, q_h);

    const dim3 blk(256);
    for (int l = 0; l < NLAY; l++) {
        // value = out @ Wv + bv  (f32 rows for the sampler)
        k_gemm_wmma<0><<<dim3(CDIM / 64, MROWS / 256), blk, 0, stream>>>(
            out_h, Wv16 + (size_t)l * CDIM * CDIM, bv + l * CDIM, CDIM,
            val_f, nullptr, MROWS, CDIM, CDIM, 0);
        // offs = q @ Wo + bo
        k_gemm_wmma<0><<<dim3(OFFC / 64, MROWS / 256), blk, 0, stream>>>(
            q_h, Wo16 + (size_t)l * CDIM * OFFC, bo + l * OFFC, OFFC,
            offs_f, nullptr, MROWS, OFFC, CDIM, 0);
        // attn logits = q @ Wa + ba (padded N=128; pad cols exactly 0)
        k_gemm_wmma<0><<<dim3(AWPAD / 64, MROWS / 256), blk, 0, stream>>>(
            q_h, Wa16 + (size_t)l * CDIM * AWPAD, ba + l * AWC, AWC,
            awl_f, nullptr, MROWS, AWPAD, CDIM, 0);

        k_sample<<<dim3(MROWS * NHEAD / 8), blk, 0, stream>>>(val_f, offs_f, awl_f, attn_h);

        // attn_out = attn @ Wout + bout
        k_gemm_wmma<0><<<dim3(CDIM / 64, MROWS / 256), blk, 0, stream>>>(
            attn_h, Wout16 + (size_t)l * CDIM * CDIM, bout + l * CDIM, CDIM,
            tmp_f, nullptr, MROWS, CDIM, CDIM, 0);
        k_resid_ln<<<dim3(MROWS / 8), blk, 0, stream>>>(
            out_f, tmp_f, ln1g + l * CDIM, ln1b + l * CDIM, out_h, nullptr, nullptr);

        // hidden = relu(out @ W1 + b1) -> f16, swizzled for K=1024 consumer
        k_gemm_wmma<1><<<dim3(DFFN / 64, MROWS / 256), blk, 0, stream>>>(
            out_h, W116 + (size_t)l * CDIM * DFFN, b1 + l * DFFN, DFFN,
            nullptr, hid_h, MROWS, DFFN, CDIM, DFFN);
        // ff = hidden @ W2 + b2
        k_gemm_wmma<0><<<dim3(CDIM / 64, MROWS / 256), blk, 0, stream>>>(
            hid_h, W216 + (size_t)l * DFFN * CDIM, b2 + l * CDIM, CDIM,
            tmp_f, nullptr, MROWS, CDIM, DFFN, 0);
        // out = LN(out + ff); fuse next layer's q = f16(out + pos)
        k_resid_ln<<<dim3(MROWS / 8), blk, 0, stream>>>(
            out_f, tmp_f, ln2g + l * CDIM, ln2b + l * CDIM, out_h, pos_f, q_h);
    }

    (void)hipMemcpyAsync(d_out, out_f, MC * sizeof(float), hipMemcpyDeviceToDevice, stream);
}
